// RNN_82532091559975
// MI455X (gfx1250) — matrix operand order repaired and checked by reading, not compiled, not run on hardware
//
#include <hip/hip_runtime.h>

// Problem constants (match reference)
#define VOCAB 401
#define E     256
#define H     768
#define G3    (3 * H)
#define O     3
#define BB    64
#define TT    1024

typedef __bf16 bf16_t;
typedef __attribute__((ext_vector_type(8)))  bf16_t v8bf;
typedef __attribute__((ext_vector_type(16))) bf16_t v16bf;
typedef __attribute__((ext_vector_type(8)))  float  v8f;

union ABFrag { v16bf v; v8bf h[2]; };

static __device__ __forceinline__ unsigned short f2bf(float f) {
  unsigned int u = __builtin_bit_cast(unsigned int, f);
  u += 0x7fffu + ((u >> 16) & 1u);   // round-to-nearest-even
  return (unsigned short)(u >> 16);
}

static __device__ __forceinline__ float sigmoid_f(float x) {
  return 1.f / (1.f + __expf(-x));
}
static __device__ __forceinline__ float tanh_f(float x) {
  // 1 - 2/(e^{2x}+1): branchless, correct limits at +/-inf
  return 1.f - 2.f / (__expf(2.f * x) + 1.f);
}

// ---------------------------------------------------------------------------
// Setup: convert weights/embedding fp32 -> bf16 (row-major, K-contiguous) and
// zero the fp32 hidden state. Fully parallel, trivially bandwidth bound.
// ---------------------------------------------------------------------------
__global__ void gru_setup(const float* __restrict__ embed,
                          const float* __restrict__ Wih,
                          const float* __restrict__ Whh,
                          unsigned short* __restrict__ wWhh,
                          unsigned short* __restrict__ wWih,
                          unsigned short* __restrict__ wEmb,
                          float* __restrict__ hf32) {
  const int n1 = G3 * H, n2 = G3 * E, n3 = VOCAB * E, n4 = BB * H;
  const int total = n1 + n2 + n3 + n4;
  for (int i = blockIdx.x * blockDim.x + threadIdx.x; i < total;
       i += gridDim.x * blockDim.x) {
    if (i < n1)                 wWhh[i]               = f2bf(Whh[i]);
    else if (i < n1 + n2)       wWih[i - n1]          = f2bf(Wih[i - n1]);
    else if (i < n1 + n2 + n3)  wEmb[i - n1 - n2]     = f2bf(embed[i - n1 - n2]);
    else                        hf32[i - n1 - n2 - n3] = 0.0f;
  }
}

// ---------------------------------------------------------------------------
// Persistent single-workgroup GRU recurrence. 1024 threads = 32 wave32 on one
// WGP. Hidden state double-buffered in LDS as bf16 (A-operand source); the
// fp32 master copy of each (row,col) is owned by exactly one lane, so it can
// live in global workspace without cross-wave races. One s_barrier per step.
// ---------------------------------------------------------------------------
__global__ __launch_bounds__(1024)
void gru_recurrent(const int* __restrict__ tokens,
                   const unsigned short* __restrict__ wWhh,   // [3H][H] bf16
                   const unsigned short* __restrict__ wWih,   // [3H][E] bf16
                   const unsigned short* __restrict__ wEmb,   // [VOCAB][E] bf16
                   const float* __restrict__ b_ih,
                   const float* __restrict__ b_hh,
                   const float* __restrict__ W_out,           // [O][H]
                   const float* __restrict__ b_out,
                   float* __restrict__ hf32,                  // [BB][H] fp32 state
                   float* __restrict__ out) {                 // [BB*O + BB*H]
  __shared__ __align__(32) unsigned short hbuf[2][BB * H];    // 192 KB (CDNA5 LDS)
  __shared__ __align__(16) float cbias[H * 4];                // fused biases, [j][4]
  __shared__ int   tokbuf[2][BB];
  __shared__ float lg[BB * O];

  const int tid    = threadIdx.x;
  const int wave   = tid >> 5;
  const int lane   = tid & 31;
  const int mtile  = wave & 3;              // M tile: 4 x 16 rows
  const int jgroup = wave >> 2;             // 8 groups x 6 column tiles
  const int lcol   = lane & 15;
  const int khalf  = (lane >> 4) << 3;      // A frag: 0 or 8; K offset 2*khalf = 0 or 16
  const int kb     = (lane >> 4) << 4;      // B frag: 0 or 16 (K offset)
  const int arow   = (mtile << 4) + lcol;   // A row this lane supplies

  // init: zero h (bf16 buffer 0), prefetch first tokens, build bias table
  for (int i = tid; i < BB * H; i += 1024) hbuf[0][i] = 0;
  if (tid < BB) tokbuf[0][tid] = tokens[tid * TT + 0];
  for (int j = tid; j < H; j += 1024) {
    cbias[j * 4 + 0] = b_ih[j]         + b_hh[j];           // r
    cbias[j * 4 + 1] = b_ih[H + j]     + b_hh[H + j];       // z
    cbias[j * 4 + 2] = b_ih[2 * H + j];                     // n (x part)
    cbias[j * 4 + 3] = b_hh[2 * H + j];                     // n (h part)
  }
  __syncthreads();

  const int mbase = (mtile << 4) + ((lane >> 4) << 3);  // C-frag row base

#pragma unroll 1
  for (int t = 0; t < TT; ++t) {
    const int cur = t & 1, nxt = cur ^ 1;
    if (tid < BB && (t + 1) < TT) tokbuf[nxt][tid] = tokens[tid * TT + (t + 1)];

    const unsigned short* hrow = &hbuf[cur][arow * H];
    const int             tk   = tokbuf[cur][arow];
    const unsigned short* xrow = wEmb + (size_t)tk * E;

#pragma unroll 1
    for (int c = 0; c < 6; ++c) {
      const int g = (jgroup * 6 + c) * 16 + lcol;            // gate column (0..767)
      const unsigned short* whr = wWhh + (size_t)g * H;      // r gate row
      const unsigned short* whz = whr + (size_t)H * H;       // z gate row
      const unsigned short* whn = whz + (size_t)H * H;       // n gate row
      const unsigned short* wir = wWih + (size_t)g * E;
      const unsigned short* wiz = wir + (size_t)H * E;
      const unsigned short* win = wiz + (size_t)H * E;

      v8f ar  = {0.f, 0.f, 0.f, 0.f, 0.f, 0.f, 0.f, 0.f};
      v8f az  = {0.f, 0.f, 0.f, 0.f, 0.f, 0.f, 0.f, 0.f};
      v8f anh = {0.f, 0.f, 0.f, 0.f, 0.f, 0.f, 0.f, 0.f};
      v8f anx = {0.f, 0.f, 0.f, 0.f, 0.f, 0.f, 0.f, 0.f};

      // hidden projection: K = 768, A from LDS, one A-frag feeds 3 gates
#pragma unroll 1
      for (int kk = 0; kk < H; kk += 32) {
        ABFrag a;
        a.h[0] = *(const v8bf*)(const void*)(hrow + kk + 2 * khalf);
        a.h[1] = *(const v8bf*)(const void*)(hrow + kk + 2 * khalf + 8);
        v16bf br = *(const v16bf*)(const void*)(whr + kk + kb);
        v16bf bz = *(const v16bf*)(const void*)(whz + kk + kb);
        v16bf bn = *(const v16bf*)(const void*)(whn + kk + kb);
        ar  = __builtin_amdgcn_wmma_f32_16x16x32_bf16(false, a.v, false, br, (short)0, ar,  false, false);
        az  = __builtin_amdgcn_wmma_f32_16x16x32_bf16(false, a.v, false, bz, (short)0, az,  false, false);
        anh = __builtin_amdgcn_wmma_f32_16x16x32_bf16(false, a.v, false, bn, (short)0, anh, false, false);
      }
      // input projection: K = 256, A = gathered embedding rows
#pragma unroll 1
      for (int kk = 0; kk < E; kk += 32) {
        ABFrag a;
        a.h[0] = *(const v8bf*)(const void*)(xrow + kk + 2 * khalf);
        a.h[1] = *(const v8bf*)(const void*)(xrow + kk + 2 * khalf + 8);
        v16bf br = *(const v16bf*)(const void*)(wir + kk + kb);
        v16bf bz = *(const v16bf*)(const void*)(wiz + kk + kb);
        v16bf bn = *(const v16bf*)(const void*)(win + kk + kb);
        ar  = __builtin_amdgcn_wmma_f32_16x16x32_bf16(false, a.v, false, br, (short)0, ar,  false, false);
        az  = __builtin_amdgcn_wmma_f32_16x16x32_bf16(false, a.v, false, bz, (short)0, az,  false, false);
        anx = __builtin_amdgcn_wmma_f32_16x16x32_bf16(false, a.v, false, bn, (short)0, anx, false, false);
      }

      // fused biases for this column: one ds_load_b128
      const float cr  = cbias[g * 4 + 0];
      const float cz  = cbias[g * 4 + 1];
      const float cnx = cbias[g * 4 + 2];
      const float cnh = cbias[g * 4 + 3];

      // gates + state update in C-fragment layout (lane: col=g, rows mbase+i)
#pragma unroll
      for (int i = 0; i < 8; ++i) {
        const float r  = sigmoid_f(ar[i] + cr);
        const float z  = sigmoid_f(az[i] + cz);
        const float nn = tanh_f(anx[i] + cnx + r * (anh[i] + cnh));
        const int   row = mbase + i;
        const float hp  = hf32[row * H + g];          // owned by this lane only
        const float hn  = (1.f - z) * nn + z * hp;
        hf32[row * H + g] = hn;
        if (t == TT - 1) {
          ((float*)hbuf)[row * H + g] = hn;           // final h fp32 -> LDS
        } else {
          hbuf[nxt][row * H + g] = f2bf(hn);
        }
      }
    }
    __syncthreads();
  }

  // ---- epilogue: h_final copy + logits + log_softmax ----
  const float* hfin = (const float*)hbuf;
  for (int i = tid; i < BB * H; i += 1024) out[BB * O + i] = hfin[i];

  if (tid < BB * O) {
    const int b = tid / O, o = tid - b * O;
    float s = b_out[o];
    const float* hr = hfin + b * H;
    const float* wr = W_out + o * H;
    for (int k = 0; k < H; ++k) s += hr[k] * wr[k];
    lg[tid] = s;
  }
  __syncthreads();
  if (tid < BB * O) {
    const int b = tid / O;
    const float l0 = lg[b * O + 0], l1 = lg[b * O + 1], l2 = lg[b * O + 2];
    const float mx  = fmaxf(l0, fmaxf(l1, l2));
    const float lse = mx + __logf(__expf(l0 - mx) + __expf(l1 - mx) + __expf(l2 - mx));
    out[tid] = lg[tid] - lse;
  }
}

// ---------------------------------------------------------------------------
extern "C" void kernel_launch(void* const* d_in, const int* in_sizes, int n_in,
                              void* d_out, int out_size, void* d_ws, size_t ws_size,
                              hipStream_t stream) {
  (void)in_sizes; (void)n_in; (void)out_size; (void)ws_size;
  const int*   tokens = (const int*)d_in[0];
  // d_in[1] = hidden (reference zeroes it; unused)
  const float* embed  = (const float*)d_in[2];
  const float* W_ih   = (const float*)d_in[3];
  const float* W_hh   = (const float*)d_in[4];
  const float* b_ih   = (const float*)d_in[5];
  const float* b_hh   = (const float*)d_in[6];
  const float* W_out  = (const float*)d_in[7];
  const float* b_out  = (const float*)d_in[8];
  float*       out    = (float*)d_out;

  // Workspace layout (≈5.1 MB): bf16 W_hh | bf16 W_ih | bf16 embed | fp32 h
  char* ws = (char*)d_ws;
  unsigned short* wWhh = (unsigned short*)ws;                 // 3H*H
  unsigned short* wWih = wWhh + (size_t)G3 * H;               // 3H*E
  unsigned short* wEmb = wWih + (size_t)G3 * E;               // VOCAB*E
  float* hf32 = (float*)(ws + ((size_t)G3 * H + (size_t)G3 * E + (size_t)VOCAB * E) * sizeof(unsigned short));

  gru_setup<<<1280, 256, 0, stream>>>(embed, W_ih, W_hh, wWhh, wWih, wEmb, hf32);
  gru_recurrent<<<1, 1024, 0, stream>>>(tokens, wWhh, wWih, wEmb,
                                        b_ih, b_hh, W_out, b_out, hf32, out);
}
